// VocosVQCodec_87265145520609
// MI455X (gfx1250) — matrix operand order repaired and checked by reading, not compiled, not run on hardware
//
#include <hip/hip_runtime.h>
#include <hip/hip_bf16.h>

// ---------------- problem constants ----------------
#define B_     16
#define D_     64
#define T_     3000
#define NCB_   8
#define K_     1024
#define N_     (B_ * T_)        // 48000 rows
#define NTILES (N_ / 16)        // 3000 row tiles (exact)
#define CTILES (K_ / 16)        // 64 code tiles
#define QSIZE  (N_ * D_)        // 3,072,000 (also == B*D*T)

// A-operand pack: [stage][ct][sel(hi/lo)][chunk][lane][16] bf16 -> 4KB per ct
#define APK_ELEMS (NCB_ * CTILES * 2 * 2 * 32 * 16)   // 1,048,576 bf16 (2 MB)
#define APK_STAGE_BYTES (CTILES * 4096)               // 256 KB per stage

#define CT_PER_CHUNK 8
#define NCHUNKS  (CTILES / CT_PER_CHUNK)              // 8
#define CHUNK_BYTES (CT_PER_CHUNK * 4096)             // 32 KB
#define LDS_BYTES (2 * CHUNK_BYTES)                   // 64 KB, ping-pong

// ---------------- workspace layout (bytes) ----------------
#define WS_R      0u                          // N_*D_ f32 residual (12,288,000 B)
#define WS_ENORM  (WS_R + (unsigned)(N_ * D_ * 4))
#define WS_USED   (WS_ENORM + (unsigned)(NCB_ * K_ * 4))
#define WS_LOSS   (WS_USED + (unsigned)(NCB_ * K_ * 4))
#define WS_APK    (WS_LOSS + 64u)             // pad to 64B; 2 MB of bf16
// total ~14.4 MB

typedef __attribute__((ext_vector_type(16))) __bf16       v16bf;
typedef __attribute__((ext_vector_type(8)))  float        v8f;
typedef __attribute__((ext_vector_type(4)))  unsigned int v4u;
typedef __attribute__((ext_vector_type(8)))  int          v8i;
typedef __attribute__((ext_vector_type(4)))  int          v4i;

// ---------------------------------------------------------------------------
// TDM: 1-D contiguous copy global -> LDS.  D# per CDNA5 ISA:
//  group0: [1:0]=count=1, [63:32]=lds_addr, [120:64]=global_addr, [127:126]=2
//  group1: [17:16]=data_size=3 (8B), [79:48]=tensor_dim0, [127:112]=tile_dim0,
//          [207:160]=tensor_dim0_stride; tile_dim1=0 => 1-D tile.
// ---------------------------------------------------------------------------
__device__ __forceinline__ void tdm_load_1d(unsigned int lds_off,
                                            const void* src,
                                            unsigned int nbytes) {
    unsigned long long ga = (unsigned long long)(uintptr_t)src;
    unsigned int units = nbytes >> 3;                 // 8-byte elements
    v4u g0;
    g0.x = 1u;                                        // count=1, user descriptor
    g0.y = lds_off;                                   // lds_addr (bytes)
    g0.z = (unsigned int)(ga & 0xffffffffu);          // global_addr[31:0]
    g0.w = (unsigned int)((ga >> 32) & 0x01ffffffu)   // global_addr[56:32]
         | (2u << 30);                                // type = 2 ("image")
    v8i g1 = (v8i)0;
    g1[0] = (int)(3u << 16);                          // data_size = 3 (8 bytes)
    g1[1] = (int)((units & 0xffffu) << 16);           // tensor_dim0[15:0]
    g1[2] = (int)((units >> 16) & 0xffffu)            // tensor_dim0[31:16]
          | (1 << 16);                                // tensor_dim1 = 1
    g1[3] = (int)((units & 0xffffu) << 16);           // tile_dim0 = units
    g1[4] = 0;                                        // tile_dim1=0, tile_dim2=0
    g1[5] = (int)units;                               // tensor_dim0_stride
    g1[6] = 0;
    g1[7] = 0;
    v4i gz = (v4i)0;
#if __clang_major__ >= 23
    v8i gz8 = (v8i)0;
    __builtin_amdgcn_tensor_load_to_lds(g0, g1, gz, gz, gz8, 0);
#else
    __builtin_amdgcn_tensor_load_to_lds(g0, g1, gz, gz, 0);
#endif
}

// ---------------------------------------------------------------------------
// Init: residual R[n,d] = x[b,d,t] (n=b*T+t); zero quantized-out, used, loss.
// ---------------------------------------------------------------------------
__global__ void vq_init_kernel(const float* __restrict__ x,
                               float* __restrict__ R,
                               float* __restrict__ out,
                               float* __restrict__ used,
                               float* __restrict__ loss) {
    int i = blockIdx.x * blockDim.x + threadIdx.x;
    if (i < QSIZE) {
        int n = i >> 6, d = i & 63;
        int b = n / T_, t = n % T_;
        R[i]   = x[((size_t)b * D_ + d) * T_ + t];
        out[i] = 0.0f;                         // quantized_total accumulator
    }
    if (i < NCB_ * K_) used[i] = 0.0f;
    if (i == 0)        loss[0] = 0.0f;
}

// ---------------------------------------------------------------------------
// Pack codebooks into WMMA A-operand lane layout (bf16 hi/lo split), grouped
// as one contiguous 4 KB block per 16-code tile so TDM can stream whole
// chunks to LDS.  16-bit A 16x32 layout (ISA 7.12.2): lane<16 holds
// K={0..7,16..23}, lane>=16 holds K={8..15,24..31}.  Also ||e_k||^2.
// ---------------------------------------------------------------------------
__global__ void vq_pack_kernel(const float* __restrict__ embeds,
                               __bf16* __restrict__ Apk,
                               float* __restrict__ enorm) {
    int gid = blockIdx.x * blockDim.x + threadIdx.x;
    if (gid < APK_ELEMS) {
        int j     = gid & 15;
        int lane  = (gid >> 4) & 31;
        int c     = (gid >> 9) & 1;
        int sel   = (gid >> 10) & 1;          // 0 = hi, 1 = lo
        int ct    = (gid >> 11) & 63;
        int stage = gid >> 17;
        int m = lane & 15, khalf = lane >> 4;
        int code = ct * 16 + m;
        int kk   = (j < 8) ? (j + khalf * 8) : (j + 8 + khalf * 8);
        int dim  = c * 32 + kk;
        float v  = embeds[((size_t)stage * K_ + code) * D_ + dim];
        __bf16 h = (__bf16)v;
        Apk[gid] = sel ? (__bf16)(v - (float)h) : h;
    } else {
        int cg = gid - APK_ELEMS;
        if (cg < NCB_ * K_) {
            const float* e = embeds + (size_t)cg * D_;
            float s = 0.0f;
            #pragma unroll
            for (int d = 0; d < D_; ++d) s += e[d] * e[d];
            enorm[cg] = s;
        }
    }
}

// ---------------------------------------------------------------------------
// One VQ stage.  8 waves/block, one 16-row tile per wave.  Wave 0 ping-pongs
// 32 KB codebook chunks into LDS with tensor_load_to_lds (TENSORcnt +
// barrier publish); all waves feed WMMA operands from LDS (ds_load_b128).
// Score tile via bf16-split WMMA, per-lane argmin over 1024 codes, then
// residual update + quantized accumulation + loss + used-flags.
// ---------------------------------------------------------------------------
__global__ __launch_bounds__(256)
void vq_stage_kernel(float* __restrict__ R,
                     const float* __restrict__ embeds,
                     const __bf16* __restrict__ Apk,
                     const float* __restrict__ enorm,
                     float* __restrict__ used,
                     float* __restrict__ loss,
                     float* __restrict__ out,
                     int stage) {
    extern __shared__ char lds_a[];                      // 2 x 32 KB ping-pong
    const int lane  = threadIdx.x & 31;
    const int wave  = threadIdx.x >> 5;
    const int tile  = blockIdx.x * 8 + wave;             // 0..2999 exact
    const int nloc  = lane & 15;
    const int khalf = lane >> 4;
    const bool leader = (threadIdx.x < 32);              // wave 0 drives TDM

    const char* apk_stage = (const char*)Apk + (size_t)stage * APK_STAGE_BYTES;
    const unsigned int lds_base = (unsigned int)(uintptr_t)&lds_a[0];

    // kick off first chunk before touching R (overlap TDM with B-operand prep)
    if (leader) tdm_load_1d(lds_base, apk_stage, CHUNK_BYTES);

    // ---- B operand: this wave's 16 rows, bf16 hi/lo, two K-chunks of 32.
    // 16-bit B 32x16 layout: lane<16 -> N=lane, K={0..7,16..23}; lane>=16 -> K={8..15,24..31}.
    v16bf bhi[2], blo[2];
    {
        const float* rrow = R + (size_t)(tile * 16 + nloc) * D_ + khalf * 8;
        #pragma unroll
        for (int c = 0; c < 2; ++c) {
            const float* srcp = rrow + c * 32;
            #pragma unroll
            for (int j = 0; j < 16; ++j) {
                float v  = srcp[(j < 8) ? j : (j + 8)];
                __bf16 h = (__bf16)v;
                bhi[c][j] = h;
                blo[c][j] = (__bf16)(v - (float)h);
            }
        }
    }

    // ---- sweep code tiles chunk-by-chunk, keep per-lane running argmin
    float bv = 3.4e38f;
    int   bi = 0;
    const float* en = enorm + stage * K_;

    for (int q = 0; q < NCHUNKS; ++q) {
        if (leader) __builtin_amdgcn_s_wait_tensorcnt(0);
        __syncthreads();   // publish chunk q; everyone done reading other buf
        if (leader && (q + 1) < NCHUNKS)
            tdm_load_1d(lds_base + (unsigned int)(((q + 1) & 1) * CHUNK_BYTES),
                        apk_stage + (size_t)(q + 1) * CHUNK_BYTES, CHUNK_BYTES);

        const char* buf = lds_a + (q & 1) * CHUNK_BYTES;
        for (int ctl = 0; ctl < CT_PER_CHUNK; ++ctl) {
            const char* tb = buf + ctl * 4096;
            v8f acc = {};
            #pragma unroll
            for (int c = 0; c < 2; ++c) {
                v16bf ah = *(const v16bf*)(tb +        c * 1024 + lane * 32);
                v16bf al = *(const v16bf*)(tb + 2048 + c * 1024 + lane * 32);
                acc = __builtin_amdgcn_wmma_f32_16x16x32_bf16(false, ah, false, bhi[c],
                                                              (short)0, acc, false, false);
                acc = __builtin_amdgcn_wmma_f32_16x16x32_bf16(false, ah, false, blo[c],
                                                              (short)0, acc, false, false);
                acc = __builtin_amdgcn_wmma_f32_16x16x32_bf16(false, al, false, bhi[c],
                                                              (short)0, acc, false, false);
            }
            // C layout: VGPR v -> code = ct*16 + khalf*8 + v, column N = lane&15
            int ct = q * CT_PER_CHUNK + ctl;
            const float* ep = en + ct * 16 + khalf * 8;
            #pragma unroll
            for (int v = 0; v < 8; ++v) {
                float score = ep[v] - 2.0f * acc[v];
                int   code  = ct * 16 + khalf * 8 + v;
                bool better = score < bv;      // strict < => lowest code wins ties
                bv = better ? score : bv;
                bi = better ? code  : bi;
            }
        }
    }

    // ---- combine the two K-halves (lane l <-> l+16)
    {
        float ov = __shfl_xor(bv, 16, 32);
        int   oi = __shfl_xor(bi, 16, 32);
        if (ov < bv || (ov == bv && oi < bi)) { bv = ov; bi = oi; }
    }
    // lanes 0..15 now hold argmin for row n = tile*16 + lane

    if (lane < 16) {
        int n = tile * 16 + lane;
        out[(size_t)QSIZE + (size_t)stage * N_ + n] = (float)bi;   // indices as f32
        used[stage * K_ + bi] = 1.0f;                              // util flag
    }

    // ---- gather q, update residual, accumulate quantized output + loss.
    // Each lane owns dims {2*lane, 2*lane+1}; unique writer per element.
    float lsum = 0.0f;
    for (int r = 0; r < 16; ++r) {
        int code = __shfl(bi, r, 32);
        int n    = tile * 16 + r;
        const float* erow = embeds + ((size_t)stage * K_ + code) * D_ + 2 * lane;
        float q0 = erow[0], q1 = erow[1];
        float* rr = R + (size_t)n * D_ + 2 * lane;
        float o0 = rr[0], o1 = rr[1];
        float d0 = o0 - q0, d1 = o1 - q1;
        rr[0] = d0; rr[1] = d1;                        // residual -= q
        int b = n / T_, t = n % T_;
        float* qo = out + (size_t)b * D_ * T_ + (size_t)(2 * lane) * T_ + t;
        qo[0]  += q0;                                  // quantized_total += q
        qo[T_] += q1;
        lsum += d0 * d0 + d1 * d1;
    }
    // wave32 reduce, one atomic per wave
    #pragma unroll
    for (int off = 16; off >= 1; off >>= 1) lsum += __shfl_xor(lsum, off, 32);
    if (lane == 0) atomicAdd(loss, lsum);
}

// ---------------------------------------------------------------------------
// Finalize scalars: commit+codebook loss = 2*mean(diff^2) per stage, /NCB;
// utilization = (#used flags)/(K*NCB).
// ---------------------------------------------------------------------------
__global__ void vq_finalize_kernel(const float* __restrict__ used,
                                   const float* __restrict__ loss,
                                   float* __restrict__ out) {
    __shared__ float sm[256];
    float s = 0.0f;
    for (int i = threadIdx.x; i < NCB_ * K_; i += 256) s += used[i];
    sm[threadIdx.x] = s;
    __syncthreads();
    for (int o = 128; o > 0; o >>= 1) {
        if (threadIdx.x < o) sm[threadIdx.x] += sm[threadIdx.x + o];
        __syncthreads();
    }
    if (threadIdx.x == 0) {
        out[(size_t)QSIZE + (size_t)NCB_ * N_]     =
            2.0f * loss[0] / (float)QSIZE / (float)NCB_;
        out[(size_t)QSIZE + (size_t)NCB_ * N_ + 1] = sm[0] / (float)(NCB_ * K_);
    }
}

// ---------------------------------------------------------------------------
extern "C" void kernel_launch(void* const* d_in, const int* in_sizes, int n_in,
                              void* d_out, int out_size, void* d_ws, size_t ws_size,
                              hipStream_t stream) {
    const float* x      = (const float*)d_in[0];   // [16,64,3000]
    const float* embeds = (const float*)d_in[1];   // [8,1024,64]
    float* out = (float*)d_out;

    char* ws = (char*)d_ws;
    float*  R     = (float*)(ws + WS_R);
    float*  enorm = (float*)(ws + WS_ENORM);
    float*  used  = (float*)(ws + WS_USED);
    float*  loss  = (float*)(ws + WS_LOSS);
    __bf16* Apk   = (__bf16*)(ws + WS_APK);

    (void)in_sizes; (void)n_in; (void)out_size; (void)ws_size;

    hipLaunchKernelGGL(vq_init_kernel, dim3((QSIZE + 255) / 256), dim3(256), 0, stream,
                       x, R, out, used, loss);
    hipLaunchKernelGGL(vq_pack_kernel,
                       dim3((APK_ELEMS + NCB_ * K_ + 255) / 256), dim3(256), 0, stream,
                       embeds, Apk, enorm);
    for (int s = 0; s < NCB_; ++s) {
        hipLaunchKernelGGL(vq_stage_kernel, dim3(NTILES / 8), dim3(256),
                           LDS_BYTES, stream,
                           R, embeds, Apk, enorm, used, loss, out, s);
    }
    hipLaunchKernelGGL(vq_finalize_kernel, dim3(1), dim3(256), 0, stream,
                       used, loss, out);
}
